// LinearformerBlock_45311904973261
// MI455X (gfx1250) — compile-verified, matrix operand order checked by reading, not run on hardware
//
#include <hip/hip_runtime.h>

// ---------------------------------------------------------------------------
// MI455X / gfx1250 Linearformer block.
// GEMMs: v_wmma_f32_16x16x32_bf16 with double-buffered LDS tiles filled by
// GLOBAL_LOAD_ASYNC_TO_LDS_B128 (ASYNCcnt-tracked async copy), pipelined so
// tile t+1 streams into LDS while tile t feeds the WMMA units.
// ---------------------------------------------------------------------------

typedef __attribute__((ext_vector_type(16))) __bf16 v16bf;
typedef __attribute__((ext_vector_type(8)))  float  v8f;

// Async global->LDS copy, 16 bytes per lane. VDST = 32-bit LDS byte offset
// (low 32 bits of the generic LDS pointer per the aperture layout),
// VADDR = 64-bit global address. Tracked with ASYNCcnt.
__device__ __forceinline__ void async_copy_b128(const void* gptr, void* lptr) {
  unsigned lds_off = (unsigned)(unsigned long long)lptr;
  asm volatile("global_load_async_to_lds_b128 %0, %1, off"
               :: "v"(lds_off), "v"(gptr) : "memory");
}
__device__ __forceinline__ void async_wait0() {
  asm volatile("s_wait_asynccnt 0x0" ::: "memory");
}

// ---------------- transpose + f32 -> bf16 convert (batched) ----------------
// out[z][c][r] = (bf16) in[zoff][r][c]
__global__ __launch_bounds__(256) void k_transpose_cvt(
    const float* __restrict__ in, __bf16* __restrict__ out,
    int ldin, int ldout,
    long inOuter, long inInner, long outBatch, int divInner)
{
  __shared__ float tile[32][33];
  int z  = blockIdx.z;
  int zo = z / divInner, zi = z - zo * divInner;
  const float* ip = in + (long)zo * inOuter + (long)zi * inInner;
  __bf16* op = out + (long)z * outBatch;

  int tx = threadIdx.x, ty = threadIdx.y;
  int c  = blockIdx.x * 32 + tx;   // input col
  int r0 = blockIdx.y * 32;        // input row base
#pragma unroll
  for (int j = 0; j < 4; ++j) {
    int r = r0 + ty + j * 8;
    tile[ty + j * 8][tx] = ip[(long)r * ldin + c];
  }
  __syncthreads();
  int oc = r0 + tx;                // output col == input row
#pragma unroll
  for (int j = 0; j < 4; ++j) {
    int orow = blockIdx.x * 32 + ty + j * 8;  // output row == input col
    op[(long)orow * ldout + oc] = (__bf16)tile[tx][ty + j * 8];
  }
}

// ------------------------------ LayerNorm ----------------------------------
__global__ __launch_bounds__(256) void k_layernorm(
    const float* __restrict__ x, const float* __restrict__ g,
    const float* __restrict__ b, __bf16* __restrict__ out, int D)
{
  __shared__ float s1[256], s2[256];
  long row = blockIdx.x;
  const float* xr = x + row * D;
  int tid = threadIdx.x;
  float v[4]; float sum = 0.f, sq = 0.f;
#pragma unroll
  for (int j = 0; j < 4; ++j) {
    v[j] = xr[tid + j * 256];
    sum += v[j]; sq += v[j] * v[j];
  }
  s1[tid] = sum; s2[tid] = sq; __syncthreads();
  for (int s = 128; s > 0; s >>= 1) {
    if (tid < s) { s1[tid] += s1[tid + s]; s2[tid] += s2[tid + s]; }
    __syncthreads();
  }
  float mu   = s1[0] / (float)D;
  float var  = s2[0] / (float)D - mu * mu;
  float rstd = rsqrtf(var + 1e-5f);
#pragma unroll
  for (int j = 0; j < 4; ++j) {
    int c = tid + j * 256;
    out[row * D + c] = (__bf16)((v[j] - mu) * rstd * g[c] + b[c]);
  }
}

// ------------------- q softmax over head dim (256-wide rows) ----------------
__global__ __launch_bounds__(256) void k_softmax_row(
    const float* __restrict__ in, __bf16* __restrict__ out, float scale)
{
  __shared__ float red[256];
  long row = blockIdx.x;
  int tid = threadIdx.x;
  float v = in[row * 256 + tid] * scale;
  red[tid] = v; __syncthreads();
  for (int s = 128; s > 0; s >>= 1) {
    if (tid < s) red[tid] = fmaxf(red[tid], red[tid + s]);
    __syncthreads();
  }
  float m = red[0]; __syncthreads();
  float e = __expf(v - m);
  red[tid] = e; __syncthreads();
  for (int s = 128; s > 0; s >>= 1) {
    if (tid < s) red[tid] += red[tid + s];
    __syncthreads();
  }
  out[row * 256 + tid] = (__bf16)(e / red[0]);
}

// ------ k softmax over sequence (columns of [B,N,D]), writing k^T bf16 ------
__global__ __launch_bounds__(256) void k_softmax_col(
    const float* __restrict__ kin, __bf16* __restrict__ kt,
    int Nseq, int D, float scale)
{
  __shared__ float red[256];
  int cidx = blockIdx.x;               // b*D + (h*DH + d)   (H*DH == D)
  int b    = cidx / D;
  int coff = cidx - b * D;
  const float* base = kin + (long)b * Nseq * D + coff;
  int tid = threadIdx.x;
  float vals[16];
  float lm = -3.4e38f;
#pragma unroll
  for (int j = 0; j < 16; ++j) {
    vals[j] = base[(long)(tid + j * 256) * D] * scale;
    lm = fmaxf(lm, vals[j]);
  }
  red[tid] = lm; __syncthreads();
  for (int s = 128; s > 0; s >>= 1) {
    if (tid < s) red[tid] = fmaxf(red[tid], red[tid + s]);
    __syncthreads();
  }
  float m = red[0]; __syncthreads();
  float ls = 0.f;
#pragma unroll
  for (int j = 0; j < 16; ++j) { vals[j] = __expf(vals[j] - m); ls += vals[j]; }
  red[tid] = ls; __syncthreads();
  for (int s = 128; s > 0; s >>= 1) {
    if (tid < s) red[tid] += red[tid + s];
    __syncthreads();
  }
  float inv = 1.0f / red[0];
#pragma unroll
  for (int j = 0; j < 16; ++j)
    kt[(long)cidx * Nseq + tid + j * 256] = (__bf16)(vals[j] * inv);
}

// ------------------------------- WMMA GEMM ---------------------------------
// C[M,N] = A[M,K] (bf16, row-major) x Bt[N,K]^T (bf16, B pre-transposed).
// 128x128 block tile, BK=32 per WMMA K-step, 8 waves each owning 32x64.
// Double-buffered LDS, filled by async global->LDS b128 copies; tile t+1
// streams in while tile t is consumed by the WMMA pipeline.
// OUTM: 0 = f32 store, 1 = bf16 store, 2 = bf16 transposed store (C^T).
template<bool BIAS, bool RES, bool GELU_, int OUTM>
__global__ __launch_bounds__(256) void k_gemm_bf16(
    const __bf16* __restrict__ A, const __bf16* __restrict__ Bt,
    void* __restrict__ Cv, const float* __restrict__ bias,
    const float* __restrict__ res,
    int K, int lda, int ldb, int ldc,
    long aOuter, long aInner, long bOuter, long bInner,
    long cOuter, long cInner, int divInner)
{
  constexpr int BMt = 128, BNt = 128, BKt = 32, LDT = 40;  // LDS row pad: 32+8
  __shared__ __bf16 sA[2][BMt * LDT];
  __shared__ __bf16 sB[2][BNt * LDT];

  int z  = blockIdx.z;
  int zo = z / divInner, zi = z - zo * divInner;
  const __bf16* Ap = A  + (long)zo * aOuter + (long)zi * aInner;
  const __bf16* Bp = Bt + (long)zo * bOuter + (long)zi * bInner;
  long cOff = (long)zo * cOuter + (long)zi * cInner;

  int tid  = threadIdx.x;
  int wid  = tid >> 5, lane = tid & 31;
  int lane16 = lane & 15, hlf = lane >> 4;
  int waveM = (wid >> 1) * 32;     // 4 wave-rows
  int waveN = (wid & 1) * 64;      // 2 wave-cols
  long blockM = (long)blockIdx.y * BMt;
  long blockN = (long)blockIdx.x * BNt;

  const __bf16* aTile = Ap + blockM * lda;
  const __bf16* bTile = Bp + blockN * ldb;

  // Per-thread staging coordinates: 2x16B chunks each for A and B.
  int c0 = tid, c1 = tid + 256;
  int r0 = c0 >> 2, kc0 = (c0 & 3) << 3;
  int r1 = c1 >> 2, kc1 = (c1 & 3) << 3;

  auto stage = [&](int kk, int buf) {
    async_copy_b128(aTile + (long)r0 * lda + kk + kc0, &sA[buf][r0 * LDT + kc0]);
    async_copy_b128(aTile + (long)r1 * lda + kk + kc1, &sA[buf][r1 * LDT + kc1]);
    async_copy_b128(bTile + (long)r0 * ldb + kk + kc0, &sB[buf][r0 * LDT + kc0]);
    async_copy_b128(bTile + (long)r1 * ldb + kk + kc1, &sB[buf][r1 * LDT + kc1]);
  };

  v8f zero = {};
  v8f acc[2][4];
#pragma unroll
  for (int mi = 0; mi < 2; ++mi)
#pragma unroll
    for (int ni = 0; ni < 4; ++ni) acc[mi][ni] = zero;

  stage(0, 0);                     // prologue: async-fill buffer 0
  int T = K / BKt;
  for (int t = 0; t < T; ++t) {
    int cur = t & 1;
    async_wait0();                 // our async copies into buf[cur] landed
    __syncthreads();               // everyone's landed; prior reads of buf[cur^1] done
    if (t + 1 < T) stage((t + 1) * BKt, cur ^ 1);   // overlap next tile

    // Per-lane fragment gathers per ISA 16-bit A/B layout:
    // K(v, half) = (v>>2)*16 + (v&3)*2 + 8*half (pairs contiguous -> b128 merges)
    v16bf af[2], bfr[4];
#pragma unroll
    for (int mi = 0; mi < 2; ++mi) {
      int row = (waveM + mi * 16 + lane16) * LDT;
#pragma unroll
      for (int v = 0; v < 8; ++v) {
        int kb = ((v >> 2) << 4) | ((v & 3) << 1) | (hlf << 3);
        af[mi][2 * v]     = sA[cur][row + kb];
        af[mi][2 * v + 1] = sA[cur][row + kb + 1];
      }
    }
#pragma unroll
    for (int ni = 0; ni < 4; ++ni) {
      int row = (waveN + ni * 16 + lane16) * LDT;
#pragma unroll
      for (int v = 0; v < 8; ++v) {
        int kb = ((v >> 2) << 4) | ((v & 3) << 1) | (hlf << 3);
        bfr[ni][2 * v]     = sB[cur][row + kb];
        bfr[ni][2 * v + 1] = sB[cur][row + kb + 1];
      }
    }
#pragma unroll
    for (int mi = 0; mi < 2; ++mi)
#pragma unroll
      for (int ni = 0; ni < 4; ++ni)
        acc[mi][ni] = __builtin_amdgcn_wmma_f32_16x16x32_bf16(
            false, af[mi], false, bfr[ni], (short)0, acc[mi][ni], false, false);

    __syncthreads();               // done reading buf[cur] before t+2 overwrites it
  }

  float*  Cf = (float*)Cv + cOff;
  __bf16* Cb = (__bf16*)Cv + cOff;
  const float* rp = RES ? (res + cOff) : nullptr;
#pragma unroll
  for (int mi = 0; mi < 2; ++mi) {
#pragma unroll
    for (int ni = 0; ni < 4; ++ni) {
#pragma unroll
      for (int v = 0; v < 8; ++v) {
        long gm = blockM + waveM + mi * 16 + v + 8 * hlf;   // D: m = v + 8*half
        long gn = blockN + waveN + ni * 16 + lane16;        //    n = lane & 15
        float val = acc[mi][ni][v];
        if (BIAS) val += bias[gn];
        if (RES)  val += rp[gm * ldc + gn];
        if (GELU_) {
          float t = val;
          val = 0.5f * t * (1.0f + tanhf(0.7978845608028654f * (t + 0.044715f * t * t * t)));
        }
        if (OUTM == 0)      Cf[gm * ldc + gn] = val;
        else if (OUTM == 1) Cb[gm * ldc + gn] = (__bf16)val;
        else                Cb[gn * ldc + gm] = (__bf16)val;   // C^T store
      }
    }
  }
}

// ---------------------------------------------------------------------------
extern "C" void kernel_launch(void* const* d_in, const int* in_sizes, int n_in,
                              void* d_out, int out_size, void* d_ws, size_t ws_size,
                              hipStream_t stream)
{
  (void)in_sizes; (void)n_in; (void)out_size; (void)ws_size;
  const int Bb = 4, Nseq = 4096, D = 1024, H = 4, DH = 256, FF = 4096;
  const long BNrows = (long)Bb * Nseq;          // 16384
  const float scale = 0.25f;                    // DH^-0.25 = 256^-0.25

  const float* x    = (const float*)d_in[0];
  const float* ln1g = (const float*)d_in[1];
  const float* ln1b = (const float*)d_in[2];
  const float* wq   = (const float*)d_in[3];
  const float* wk   = (const float*)d_in[4];
  const float* wv   = (const float*)d_in[5];
  const float* wo   = (const float*)d_in[6];
  const float* bo   = (const float*)d_in[7];
  const float* ln2g = (const float*)d_in[8];
  const float* ln2b = (const float*)d_in[9];
  const float* wff1 = (const float*)d_in[10];
  const float* bff1 = (const float*)d_in[11];
  const float* wff2 = (const float*)d_in[12];
  const float* bff2 = (const float*)d_in[13];
  const float* wd   = (const float*)d_in[14];
  const float* bd   = (const float*)d_in[15];
  float* out = (float*)d_out;

  // -------- workspace layout (byte offsets, regions reused when dead) -------
  char* ws = (char*)d_ws;
  const size_t MBy = 1ull << 20;
  __bf16* wqT  = (__bf16*)(ws + 0 * MBy);     // 2 MiB each (1024x1024 bf16)
  __bf16* wkT  = (__bf16*)(ws + 2 * MBy);
  __bf16* wvT  = (__bf16*)(ws + 4 * MBy);
  __bf16* woT  = (__bf16*)(ws + 6 * MBy);
  __bf16* wdT  = (__bf16*)(ws + 8 * MBy);
  __bf16* ff1T = (__bf16*)(ws + 10 * MBy);    // 8 MiB (4096x1024)
  __bf16* ff2T = (__bf16*)(ws + 18 * MBy);    // 8 MiB (1024x4096)
  __bf16* hbf  = (__bf16*)(ws + 26 * MBy);    // 32 MiB [16384,1024]
  float*  qf   = (float*) (ws + 58 * MBy);    // 64 MiB
  float*  kf   = (float*) (ws + 122 * MBy);   // 64 MiB
  float*  vf   = (float*) (ws + 186 * MBy);   // 64 MiB
  __bf16* qs   = (__bf16*)(ws + 250 * MBy);   // 32 MiB [B,N,D]
  __bf16* kt   = (__bf16*)(ws + 282 * MBy);   // 32 MiB [B,H,DH,N]
  __bf16* vt   = (__bf16*)(ws + 314 * MBy);   // 32 MiB [B,H,DH,N]
  __bf16* ctxT = (__bf16*)(ws + 346 * MBy);   // 2 MiB  [16,DH,DH] = ctx^T
  __bf16* attn = (__bf16*)(ws + 348 * MBy);   // 32 MiB [B,N,D]
  float*  x2   = (float*) (ws + 380 * MBy);   // 64 MiB
  __bf16* h2   = hbf;                         // reuse (hbf dead after QKV)
  __bf16* ffh  = (__bf16*)qf;                 // reuse q+k (128 MiB, [16384,4096])
  __bf16* x3b  = (__bf16*)vf;                 // reuse v (32 MiB)

  dim3 blk1(256), blkT(32, 8);

  // ---- 1) weight convert + transpose to bf16 W^T --------------------------
  k_transpose_cvt<<<dim3(32, 32, 1), blkT, 0, stream>>>(wq,  wqT, 1024, 1024, 0, 0, 0, 1);
  k_transpose_cvt<<<dim3(32, 32, 1), blkT, 0, stream>>>(wk,  wkT, 1024, 1024, 0, 0, 0, 1);
  k_transpose_cvt<<<dim3(32, 32, 1), blkT, 0, stream>>>(wv,  wvT, 1024, 1024, 0, 0, 0, 1);
  k_transpose_cvt<<<dim3(32, 32, 1), blkT, 0, stream>>>(wo,  woT, 1024, 1024, 0, 0, 0, 1);
  k_transpose_cvt<<<dim3(32, 32, 1), blkT, 0, stream>>>(wd,  wdT, 1024, 1024, 0, 0, 0, 1);
  k_transpose_cvt<<<dim3(128, 32, 1), blkT, 0, stream>>>(wff1, ff1T, 4096, 1024, 0, 0, 0, 1);
  k_transpose_cvt<<<dim3(32, 128, 1), blkT, 0, stream>>>(wff2, ff2T, 1024, 4096, 0, 0, 0, 1);

  // ---- 2) LN1 -> bf16 ------------------------------------------------------
  k_layernorm<<<dim3((unsigned)BNrows), blk1, 0, stream>>>(x, ln1g, ln1b, hbf, D);

  // ---- 3) QKV GEMMs: [16384,1024] x [1024,1024] -> f32 ---------------------
  dim3 gQKV(8, 128, 1);
  k_gemm_bf16<false, false, false, 0><<<gQKV, blk1, 0, stream>>>(
      hbf, wqT, qf, nullptr, nullptr, 1024, 1024, 1024, 1024, 0, 0, 0, 0, 0, 0, 1);
  k_gemm_bf16<false, false, false, 0><<<gQKV, blk1, 0, stream>>>(
      hbf, wkT, kf, nullptr, nullptr, 1024, 1024, 1024, 1024, 0, 0, 0, 0, 0, 0, 1);
  k_gemm_bf16<false, false, false, 0><<<gQKV, blk1, 0, stream>>>(
      hbf, wvT, vf, nullptr, nullptr, 1024, 1024, 1024, 1024, 0, 0, 0, 0, 0, 0, 1);

  // ---- 4) q softmax over DH; 5) k softmax over N -> k^T; 6) v -> v^T -------
  k_softmax_row<<<dim3(65536), blk1, 0, stream>>>(qf, qs, scale);
  k_softmax_col<<<dim3(4096), blk1, 0, stream>>>(kf, kt, Nseq, D, scale);
  k_transpose_cvt<<<dim3(8, 128, 16), blkT, 0, stream>>>(
      vf, vt, D, Nseq, (long)Nseq * D, DH, (long)DH * Nseq, H);

  // ---- 7) context^T = (k^T v)^T per (b,h): M=N=256, K=4096 -----------------
  k_gemm_bf16<false, false, false, 2><<<dim3(2, 2, 16), blk1, 0, stream>>>(
      kt, vt, ctxT, nullptr, nullptr, Nseq, Nseq, Nseq, DH,
      (long)DH * Nseq, 0, (long)DH * Nseq, 0, (long)DH * DH, 0, 1);

  // ---- 8) attn = q_s @ ctx per (b,h): M=4096, N=256, K=256 -----------------
  k_gemm_bf16<false, false, false, 1><<<dim3(2, 32, 16), blk1, 0, stream>>>(
      qs, ctxT, attn, nullptr, nullptr, DH, D, DH, D,
      (long)Nseq * D, DH, (long)H * DH * DH, (long)DH * DH,
      (long)Nseq * D, DH, H);

  // ---- 9) x2 = x + attn @ wo + bo ------------------------------------------
  k_gemm_bf16<true, true, false, 0><<<gQKV, blk1, 0, stream>>>(
      attn, woT, x2, bo, x, 1024, 1024, 1024, 1024, 0, 0, 0, 0, 0, 0, 1);

  // ---- 10) LN2 -> bf16 -----------------------------------------------------
  k_layernorm<<<dim3((unsigned)BNrows), blk1, 0, stream>>>(x2, ln2g, ln2b, h2, D);

  // ---- 11) ffh = gelu(h2 @ w_ff1 + b_ff1) -> bf16 --------------------------
  k_gemm_bf16<true, false, true, 1><<<dim3(32, 128, 1), blk1, 0, stream>>>(
      h2, ff1T, ffh, bff1, nullptr, 1024, 1024, 1024, FF, 0, 0, 0, 0, 0, 0, 1);

  // ---- 12) x3 = x2 + ffh @ w_ff2 + b_ff2 -> bf16 ---------------------------
  k_gemm_bf16<true, true, false, 1><<<dim3(8, 128, 1), blk1, 0, stream>>>(
      ffh, ff2T, x3b, bff2, x2, FF, FF, FF, 1024, 0, 0, 0, 0, 0, 0, 1);

  // ---- 13) out = x3 @ w_d + b_d (f32) --------------------------------------
  k_gemm_bf16<true, false, false, 0><<<dim3(8, 128, 1), blk1, 0, stream>>>(
      x3b, wdT, out, bd, nullptr, 1024, 1024, 1024, 1024, 0, 0, 0, 0, 0, 0, 1);
}
